// TransformerBlock_4398046511150
// MI455X (gfx1250) — matrix operand order repaired and checked by reading, not compiled, not run on hardware
//
#include <hip/hip_runtime.h>
#include <math.h>

typedef _Float16 h8  __attribute__((ext_vector_type(8)));
typedef _Float16 h16 __attribute__((ext_vector_type(16)));
typedef float    f8  __attribute__((ext_vector_type(8)));

#define BB   8
#define CC   256
#define HID  512
#define NPIX 25600   // 160*160
#define SPLK 16      // split-K chunks for the scores GEMM
#define KCHK (NPIX / SPLK)   // 1600

// ---------------------------------------------------------------------------
// Register-blocked WMMA core: MT x NT grid of 16x16 D-tiles per wave.
// A (row-major, M x K) and BT (row-major, N x K) fragment layouts follow
// CDNA5 ISA 7.12.2:
//   A: lane(0..15)=row M, VGPR0..3 K=koff..koff+7, VGPR4..7 K=16+koff  (koff=8*(lane>=16))
//   B: lane&15 = col N, same K order as A: K=koff..koff+7, then K=16+koff..16+koff+7
// 12 b128 loads feed MT*NT wmmas per k-step (vs 4:1 unblocked).
// ---------------------------------------------------------------------------
template<int MT, int NT>
__device__ __forceinline__ void wmma_block(const _Float16* __restrict__ A, size_t lda,
                                           const _Float16* __restrict__ BT, size_t ldbt,
                                           int K, int lane, f8 (&acc)[MT][NT])
{
    const int rowA  = lane & 15;
    const int koffA = (lane >> 4) << 3;   // 0 or 8
    const int colB  = lane & 15;
    const int koffB = (lane >> 4) << 3;   // 0 or 8
    const _Float16* pa = A  + (size_t)rowA * lda  + koffA;
    const _Float16* pb = BT + (size_t)colB * ldbt + koffB;
    for (int k0 = 0; k0 < K; k0 += 32) {
        h16 af[MT], bf[NT];
#pragma unroll
        for (int i = 0; i < MT; ++i) {
            const _Float16* p = pa + (size_t)i * 16 * lda + k0;
            h8 lo = *(const h8*)p;
            h8 hi = *(const h8*)(p + 16);
#pragma unroll
            for (int c = 0; c < 8; ++c) { af[i][c] = lo[c]; af[i][c + 8] = hi[c]; }
        }
#pragma unroll
        for (int j = 0; j < NT; ++j) {
            const _Float16* p = pb + (size_t)j * 16 * ldbt + k0;
            h8 lo = *(const h8*)p;
            h8 hi = *(const h8*)(p + 16);
#pragma unroll
            for (int c = 0; c < 8; ++c) { bf[j][c] = lo[c]; bf[j][c + 8] = hi[c]; }
        }
#pragma unroll
        for (int i = 0; i < MT; ++i)
#pragma unroll
            for (int j = 0; j < NT; ++j)
                acc[i][j] = __builtin_amdgcn_wmma_f32_16x16x32_f16(
                    false, af[i], false, bf[j], (short)0, acc[i][j], false, false);
    }
}

// D layout (ISA 7.12.2): VGPR r -> M = r + 8*(lane>=16), N = lane&15.
__device__ __forceinline__ void d_coords(int lane, int& nl, int& mo)
{
    nl = lane & 15;
    mo = (lane >> 4) << 3;
}

// ---------------------------------------------------------------------------
// f32 -> f16 weight conversion
// ---------------------------------------------------------------------------
__global__ void k_cvt_f16(const float* __restrict__ src, _Float16* __restrict__ dst, int n)
{
    int i = blockIdx.x * blockDim.x + threadIdx.x;
    if (i < n) dst[i] = (_Float16)src[i];
}

// ---------------------------------------------------------------------------
// Channel LayerNorm (over C=256) of BCHW f32 -> pixel-major (N x C) f16.
// One block = 32 consecutive pixels of one batch; LDS transpose gives
// coalesced global loads (along n) and coalesced f16 stores (along c).
// ---------------------------------------------------------------------------
__global__ void k_ln_f16T(const float* __restrict__ X, const float* __restrict__ g,
                          const float* __restrict__ be, _Float16* __restrict__ XT)
{
    __shared__ float tile[CC * 33];        // [c][n] padded
    __shared__ float rs[8][32], rs2[8][32];
    __shared__ float mean[32], rstd[32];

    const int t  = threadIdx.x;            // 256 threads
    const int nl = t & 31;
    const int cs = t >> 5;                 // 0..7
    const int b  = blockIdx.x / (NPIX / 32);
    const int n0 = (blockIdx.x % (NPIX / 32)) * 32;

    const float* xb = X + ((size_t)b * CC) * NPIX + n0 + nl;
    float s = 0.f, s2 = 0.f;
#pragma unroll 4
    for (int i = 0; i < 32; ++i) {
        int c = cs * 32 + i;
        float v = xb[(size_t)c * NPIX];
        tile[c * 33 + nl] = v;
        s += v; s2 += v * v;
    }
    rs[cs][nl] = s; rs2[cs][nl] = s2;
    __syncthreads();
    if (cs == 0) {
        float S = 0.f, S2 = 0.f;
#pragma unroll
        for (int j = 0; j < 8; ++j) { S += rs[j][nl]; S2 += rs2[j][nl]; }
        float mu  = S * (1.0f / CC);
        float var = S2 * (1.0f / CC) - mu * mu;
        mean[nl] = mu;
        rstd[nl] = rsqrtf(var + 1e-5f);
    }
    __syncthreads();

    const float gc = g[t], bc = be[t];     // c == t in phase 2
    _Float16* out = XT + ((size_t)b * NPIX + n0) * CC + t;
#pragma unroll 4
    for (int n = 0; n < 32; ++n) {
        float v = (tile[t * 33 + n] - mean[n]) * rstd[n] * gc + bc;
        out[(size_t)n * CC] = (_Float16)v;
    }
}

// ---------------------------------------------------------------------------
// qkv = qkv_w(768x256) x xn(256xN) + b.  64x32 macro-tile per wave.
// q,k stored channel-major (CxN) f16; v stored pixel-major (NxC) f16.
// Region boundaries (0/256/512) are multiples of 64 -> uniform per macro-tile.
// ---------------------------------------------------------------------------
__global__ void __launch_bounds__(256)
k_qkv(const _Float16* __restrict__ Wh, const float* __restrict__ bias,
      const _Float16* __restrict__ XN,
      _Float16* __restrict__ Qb, _Float16* __restrict__ Kb, _Float16* __restrict__ Vt)
{
    const int b = blockIdx.y;
    const int TMB = 768 / 64, TNB = NPIX / 32;      // 12 x 800
    int w = blockIdx.x * 8 + (threadIdx.x >> 5);
    if (w >= TMB * TNB) return;                     // wave-uniform
    int tmb = w % TMB, tnb = w / TMB;
    int lane = threadIdx.x & 31;
    int m0 = tmb * 64, n0 = tnb * 32;

    const _Float16* A  = Wh + (size_t)m0 * CC;
    const _Float16* BT = XN + (size_t)b * NPIX * CC + (size_t)n0 * CC;
    f8 acc[4][2] = {};
    wmma_block<4, 2>(A, CC, BT, CC, CC, lane, acc);

    int nl, mo; d_coords(lane, nl, mo);
    if (m0 < 2 * CC) {                              // q or k region: C x N
        _Float16* dst = (m0 < CC) ? (Qb + (size_t)b * CC * NPIX)
                                  : (Kb + (size_t)b * CC * NPIX);
        int rb = (m0 & (CC - 1));
#pragma unroll
        for (int i = 0; i < 4; ++i)
#pragma unroll
            for (int j = 0; j < 2; ++j)
#pragma unroll
                for (int r = 0; r < 8; ++r)
                    dst[(size_t)(rb + i * 16 + mo + r) * NPIX + n0 + j * 16 + nl] =
                        (_Float16)(acc[i][j][r] + bias[m0 + i * 16 + mo + r]);
    } else {                                        // v region: transposed N x C
#pragma unroll
        for (int i = 0; i < 4; ++i)
#pragma unroll
            for (int j = 0; j < 2; ++j) {
                _Float16* dst = Vt + (size_t)b * NPIX * CC +
                                (size_t)(n0 + j * 16 + nl) * CC +
                                (m0 - 2 * CC) + i * 16 + mo;
                h8 pk;
#pragma unroll
                for (int r = 0; r < 8; ++r)
                    pk[r] = (_Float16)(acc[i][j][r] + bias[m0 + i * 16 + mo + r]);
                *(h8*)dst = pk;
            }
    }
}

// ---------------------------------------------------------------------------
// scores partials: SCP[ch] = q[:, ch-chunk] x k[:, ch-chunk]^T
// 32x32 macro-tile per wave, deterministic split-K over 16 chunks of 1600.
// ---------------------------------------------------------------------------
__global__ void __launch_bounds__(256)
k_scores(const _Float16* __restrict__ Qb, const _Float16* __restrict__ Kb,
         float* __restrict__ SCP)
{
    const int b = blockIdx.y;
    int w = blockIdx.x * 8 + (threadIdx.x >> 5);    // 8*8*16 = 1024 wave-jobs
    int tmb = w & 7, tnb = (w >> 3) & 7, ch = w >> 6;
    int lane = threadIdx.x & 31;

    const _Float16* A  = Qb + (size_t)b * CC * NPIX + (size_t)tmb * 32 * NPIX + (size_t)ch * KCHK;
    const _Float16* BT = Kb + (size_t)b * CC * NPIX + (size_t)tnb * 32 * NPIX + (size_t)ch * KCHK;
    f8 acc[2][2] = {};
    wmma_block<2, 2>(A, NPIX, BT, NPIX, KCHK, lane, acc);

    int nl, mo; d_coords(lane, nl, mo);
    float* out = SCP + ((size_t)ch * BB + b) * CC * CC;
#pragma unroll
    for (int i = 0; i < 2; ++i)
#pragma unroll
        for (int j = 0; j < 2; ++j)
#pragma unroll
            for (int r = 0; r < 8; ++r)
                out[(size_t)(tmb * 32 + i * 16 + mo + r) * CC + tnb * 32 + j * 16 + nl] =
                    acc[i][j][r];
}

// ---------------------------------------------------------------------------
// split-K reduction fused with softmax (1/sqrt(C)=1/16 prescale).
// One wave per 256-wide row; wave32 shuffle reductions.
// ---------------------------------------------------------------------------
__global__ void k_softmax(const float* __restrict__ SCP, float* __restrict__ SC)
{
    int row  = blockIdx.x * 8 + (threadIdx.x >> 5); // 0 .. B*256-1
    int b    = row >> 8;
    int r    = row & 255;
    int lane = threadIdx.x & 31;

    f8 v = (f8)0.0f;
#pragma unroll
    for (int ch = 0; ch < SPLK; ++ch)
        v += *(const f8*)(SCP + (((size_t)ch * BB + b) * CC + r) * CC + lane * 8);

    float sv[8], mx = -3.4e38f;
#pragma unroll
    for (int i = 0; i < 8; ++i) { sv[i] = v[i] * 0.0625f; mx = fmaxf(mx, sv[i]); }
#pragma unroll
    for (int off = 16; off >= 1; off >>= 1) mx = fmaxf(mx, __shfl_xor(mx, off));
    float s = 0.f;
#pragma unroll
    for (int i = 0; i < 8; ++i) { sv[i] = __expf(sv[i] - mx); s += sv[i]; }
#pragma unroll
    for (int off = 16; off >= 1; off >>= 1) s += __shfl_xor(s, off);
    float inv = 1.0f / s;
#pragma unroll
    for (int i = 0; i < 8; ++i) v[i] = sv[i] * inv;
    *(f8*)(SC + (size_t)row * CC + lane * 8) = v;
}

// ---------------------------------------------------------------------------
// M = proj_w(256x256) @ attn(256x256) (tiny f32 GEMM -> f16 row-major).
// Fuses the proj 1x1-conv into the attn matrix: proj(attn@v) == (proj@attn)@v
// ---------------------------------------------------------------------------
__global__ void k_fuseproj(const float* __restrict__ Pw, const float* __restrict__ SC,
                           _Float16* __restrict__ Mh)
{
    int b = blockIdx.y, m = blockIdx.x, d = threadIdx.x;
    const float* att = SC + (size_t)b * CC * CC;
    float acc = 0.f;
#pragma unroll 4
    for (int c = 0; c < CC; ++c) acc += Pw[m * CC + c] * att[c * CC + d];
    Mh[(size_t)b * CC * CC + m * CC + d] = (_Float16)acc;
}

// ---------------------------------------------------------------------------
// x1 = x + M(256x256) @ v(256xN) + proj_b   -> d_out (f32 BCHW)
// ---------------------------------------------------------------------------
__global__ void __launch_bounds__(256)
k_attnv(const _Float16* __restrict__ Mh, const _Float16* __restrict__ Vt,
        const float* __restrict__ bias, const float* __restrict__ Xin,
        float* __restrict__ Out)
{
    const int b = blockIdx.y;
    const int TMB = CC / 64, TNB = NPIX / 32;       // 4 x 800
    int w = blockIdx.x * 8 + (threadIdx.x >> 5);
    if (w >= TMB * TNB) return;
    int tmb = w % TMB, tnb = w / TMB;
    int lane = threadIdx.x & 31;
    int m0 = tmb * 64, n0 = tnb * 32;

    const _Float16* A  = Mh + (size_t)b * CC * CC + (size_t)m0 * CC;
    const _Float16* BT = Vt + (size_t)b * NPIX * CC + (size_t)n0 * CC;
    f8 acc[4][2] = {};
    wmma_block<4, 2>(A, CC, BT, CC, CC, lane, acc);

    int nl, mo; d_coords(lane, nl, mo);
#pragma unroll
    for (int i = 0; i < 4; ++i)
#pragma unroll
        for (int j = 0; j < 2; ++j)
#pragma unroll
            for (int r = 0; r < 8; ++r) {
                int m = m0 + i * 16 + mo + r;
                size_t idx = ((size_t)b * CC + m) * NPIX + n0 + j * 16 + nl;
                Out[idx] = Xin[idx] + acc[i][j][r] + bias[m];
            }
}

// ---------------------------------------------------------------------------
// h = gelu_exact(w1(512x256) @ yn(256xN) + b1) -> f16 transposed (N x 512)
// ---------------------------------------------------------------------------
__global__ void __launch_bounds__(256)
k_ffn1(const _Float16* __restrict__ W1h, const float* __restrict__ b1,
       const _Float16* __restrict__ XN, _Float16* __restrict__ H)
{
    const int b = blockIdx.y;
    const int TMB = HID / 64, TNB = NPIX / 32;      // 8 x 800
    int w = blockIdx.x * 8 + (threadIdx.x >> 5);
    if (w >= TMB * TNB) return;
    int tmb = w % TMB, tnb = w / TMB;
    int lane = threadIdx.x & 31;
    int m0 = tmb * 64, n0 = tnb * 32;

    const _Float16* A  = W1h + (size_t)m0 * CC;
    const _Float16* BT = XN + (size_t)b * NPIX * CC + (size_t)n0 * CC;
    f8 acc[4][2] = {};
    wmma_block<4, 2>(A, CC, BT, CC, CC, lane, acc);

    int nl, mo; d_coords(lane, nl, mo);
#pragma unroll
    for (int i = 0; i < 4; ++i)
#pragma unroll
        for (int j = 0; j < 2; ++j) {
            _Float16* dst = H + (size_t)b * NPIX * HID +
                            (size_t)(n0 + j * 16 + nl) * HID + m0 + i * 16 + mo;
            h8 pk;
#pragma unroll
            for (int r = 0; r < 8; ++r) {
                float x = acc[i][j][r] + b1[m0 + i * 16 + mo + r];
                x = 0.5f * x * (1.0f + erff(x * 0.70710678118f));   // exact gelu
                pk[r] = (_Float16)x;
            }
            *(h8*)dst = pk;
        }
}

// ---------------------------------------------------------------------------
// out = x1 + w2(256x512) @ h(512xN) + b2   (in place on d_out)
// ---------------------------------------------------------------------------
__global__ void __launch_bounds__(256)
k_ffn2(const _Float16* __restrict__ W2h, const float* __restrict__ b2,
       const _Float16* __restrict__ H, float* __restrict__ Out)
{
    const int b = blockIdx.y;
    const int TMB = CC / 64, TNB = NPIX / 32;       // 4 x 800
    int w = blockIdx.x * 8 + (threadIdx.x >> 5);
    if (w >= TMB * TNB) return;
    int tmb = w % TMB, tnb = w / TMB;
    int lane = threadIdx.x & 31;
    int m0 = tmb * 64, n0 = tnb * 32;

    const _Float16* A  = W2h + (size_t)m0 * HID;
    const _Float16* BT = H + (size_t)b * NPIX * HID + (size_t)n0 * HID;
    f8 acc[4][2] = {};
    wmma_block<4, 2>(A, HID, BT, HID, HID, lane, acc);

    int nl, mo; d_coords(lane, nl, mo);
#pragma unroll
    for (int i = 0; i < 4; ++i)
#pragma unroll
        for (int j = 0; j < 2; ++j)
#pragma unroll
            for (int r = 0; r < 8; ++r) {
                int m = m0 + i * 16 + mo + r;
                size_t idx = ((size_t)b * CC + m) * NPIX + n0 + j * 16 + nl;
                Out[idx] = Out[idx] + acc[i][j][r] + b2[m];
            }
}

// ---------------------------------------------------------------------------
extern "C" void kernel_launch(void* const* d_in, const int* in_sizes, int n_in,
                              void* d_out, int out_size, void* d_ws, size_t ws_size,
                              hipStream_t stream)
{
    const float* x      = (const float*)d_in[0];
    const float* ln1_w  = (const float*)d_in[1];
    const float* ln1_b  = (const float*)d_in[2];
    const float* qkv_w  = (const float*)d_in[3];
    const float* qkv_b  = (const float*)d_in[4];
    const float* proj_w = (const float*)d_in[5];
    const float* proj_b = (const float*)d_in[6];
    const float* ln2_w  = (const float*)d_in[7];
    const float* ln2_b  = (const float*)d_in[8];
    const float* w1     = (const float*)d_in[9];
    const float* b1     = (const float*)d_in[10];
    const float* w2     = (const float*)d_in[11];
    const float* b2     = (const float*)d_in[12];
    float* out = (float*)d_out;

    // workspace carve-up (f16 unless noted)
    _Float16* XN  = (_Float16*)d_ws;                       // B x N x C  (xn, later yn)
    _Float16* Qb  = XN + (size_t)BB * NPIX * CC;           // B x C x N
    _Float16* Kb  = Qb + (size_t)BB * CC * NPIX;           // B x C x N
    _Float16* Vt  = Kb + (size_t)BB * CC * NPIX;           // B x N x C
    float*    SC  = (float*)(Vt + (size_t)BB * NPIX * CC); // B x C x C f32 (attn)
    _Float16* Mh  = (_Float16*)(SC + (size_t)BB * CC * CC);// B x C x C
    _Float16* Wqh = Mh + (size_t)BB * CC * CC;             // 768x256
    _Float16* W1h = Wqh + 768 * CC;                        // 512x256
    _Float16* W2h = W1h + HID * CC;                        // 256x512
    float*    SCP = (float*)(W2h + CC * HID);              // SPLK x B x C x C f32 partials
    _Float16* H   = Qb;          // reuse q/k region (dead after scores) for B x N x HID

    // weight conversion
    k_cvt_f16<<<(768 * CC + 255) / 256, 256, 0, stream>>>(qkv_w, Wqh, 768 * CC);
    k_cvt_f16<<<(HID * CC + 255) / 256, 256, 0, stream>>>(w1, W1h, HID * CC);
    k_cvt_f16<<<(CC * HID + 255) / 256, 256, 0, stream>>>(w2, W2h, CC * HID);

    dim3 blk(256);
    // LN1 -> xn (N x C f16)
    k_ln_f16T<<<dim3(BB * NPIX / 32), blk, 0, stream>>>(x, ln1_w, ln1_b, XN);
    // qkv GEMM (12*800 macro-tiles / 8 waves per block)
    k_qkv<<<dim3(12 * (NPIX / 32) / 8, BB), blk, 0, stream>>>(Wqh, qkv_b, XN, Qb, Kb, Vt);
    // channel-attention score partials, deterministic split-K
    k_scores<<<dim3(8 * 8 * SPLK / 8, BB), blk, 0, stream>>>(Qb, Kb, SCP);
    // split-K reduce + softmax -> SC
    k_softmax<<<dim3(BB * CC / 8), blk, 0, stream>>>(SCP, SC);
    // M = proj_w @ attn (proj fused into attention matrix)
    k_fuseproj<<<dim3(CC, BB), blk, 0, stream>>>(proj_w, SC, Mh);
    // x1 = x + M @ v + proj_b  -> d_out
    k_attnv<<<dim3(4 * (NPIX / 32) / 8, BB), blk, 0, stream>>>(Mh, Vt, proj_b, x, out);
    // LN2 -> yn (reuses XN)
    k_ln_f16T<<<dim3(BB * NPIX / 32), blk, 0, stream>>>(out, ln2_w, ln2_b, XN);
    // h = gelu(w1 @ yn + b1) -> H (N x 512 f16, overwrites dead q/k)
    k_ffn1<<<dim3(8 * (NPIX / 32) / 8, BB), blk, 0, stream>>>(W1h, b1, XN, H);
    // out = x1 + w2 @ h + b2
    k_ffn2<<<dim3(4 * (NPIX / 32) / 8, BB), blk, 0, stream>>>(W2h, b2, H, out);
}
